// BiLSTMStack_17987323035827
// MI455X (gfx1250) — compile-verified, matrix operand order checked by reading, not run on hardware
//
#include <hip/hip_runtime.h>
#include <hip/hip_bf16.h>
#include <stdint.h>

// ---------------------------------------------------------------------------
// BiLSTM stack for MI455X (gfx1250, wave32, WMMA bf16 16x16x32)
//   block b: xs[T,B,C] -> fwd/bwd LSTM(H=128) -> concat -> 1x1 conv -> BN
// Strategy:
//   * all GEMMs via v_wmma_f32_16x16x32_bf16 (f32 accumulate)
//   * persistent recurrent kernel, templated on C (no inner-loop branches):
//       - Whh fragments live in VGPRs for the whole 3000-step scan
//       - gate biases live in VGPRs as persistent C-operand tiles
//         (first WMMA of each gate consumes them: zero init-movs per step)
//       - Wih lives in LDS; hidden state double-buffered in LDS as bf16
//       - strength-reduced x/y pointers (no per-step 64-bit addr math)
//       - cell state in VGPRs; 8 workgroups = 4 batch tiles x 2 directions
//   * conv 1x1 as a big parallel WMMA GEMM (weights staged in LDS)
//   * BatchNorm: deterministic 2-stage reduction (no float atomics)
// ---------------------------------------------------------------------------

#define T_LEN  3000
#define NBATCH 64
#define HID    128
#define GATES  512
#define TBROWS (T_LEN * NBATCH)      // 192000
#define MTILES (TBROWS / 16)         // 12000

typedef __attribute__((ext_vector_type(16))) __bf16 v16bf;
typedef __attribute__((ext_vector_type(8)))  float  v8f;

__device__ __forceinline__ unsigned short f2bf(float f) {
    unsigned u = __float_as_uint(f);
    unsigned r = (u + 0x7FFFu + ((u >> 16) & 1u)) >> 16;   // RNE
    return (unsigned short)r;
}

__device__ __forceinline__ float hsig(float x) {
    return fminf(fmaxf(__builtin_fmaf(0.2f, x, 0.5f), 0.0f), 1.0f);
}

union FragU { v16bf v; uint4 q[2]; };

// Load a 16x32 bf16 A/B fragment. base points at element [row0, kbase] of a
// row-major bf16 matrix with row stride `stride` (elements). Per ISA 7.12.2:
// lane<16 -> row l, K {0..7, 16..23}; lane>=16 -> row l, K {8..15, 24..31}.
__device__ __forceinline__ v16bf load_frag(const unsigned short* base, int stride, int lane) {
    const int l  = lane & 15;
    const int hi = (lane >> 4) & 1;
    const unsigned short* p = base + l * stride + hi * 8;
    FragU u;
    u.q[0] = *(const uint4*)(p);
    u.q[1] = *(const uint4*)(p + 16);
    return u.v;
}

__device__ __forceinline__ v8f wmma_bf16(v16bf a, v16bf b, v8f c) {
    return __builtin_amdgcn_wmma_f32_16x16x32_bf16(false, a, false, b, (short)0, c, false, false);
}

// ---------------------------------------------------------------------------
// x [B=64][C=64][T] f32  ->  Xbf [(t*64+b)][c] bf16 (row stride 64)
__global__ void transform_x(const float* __restrict__ x, unsigned short* __restrict__ Xbf) {
    size_t i = (size_t)blockIdx.x * blockDim.x + threadIdx.x;   // < TBROWS*64
    int c   = (int)(i & 63);
    size_t row = i >> 6;
    int b = (int)(row & 63);
    int t = (int)(row >> 6);
    Xbf[i] = f2bf(x[((size_t)b * 64 + c) * T_LEN + t]);
}

__global__ void cvt_bf16(const float* __restrict__ src, unsigned short* __restrict__ dst, int n) {
    int i = blockIdx.x * blockDim.x + threadIdx.x;
    if (i < n) dst[i] = f2bf(src[i]);
}

__global__ void add_f32(const float* __restrict__ a, const float* __restrict__ b,
                        float* __restrict__ dst, int n) {
    int i = blockIdx.x * blockDim.x + threadIdx.x;
    if (i < n) dst[i] = a[i] + b[i];
}

// ---------------------------------------------------------------------------
// Persistent bidirectional LSTM scan.
// grid.x = 8 : (blockIdx.x & 3) = batch 16-row tile, (blockIdx.x >> 2) = dir
// block = 256 threads (8 waves); wave w owns hidden column group hg = w.
// LDS: Wih[512][C] bf16 | h double buffer 2x[16][128] bf16
extern __shared__ unsigned short lds_lstm[];

template <int C>
__global__ void __launch_bounds__(256)
lstm_persistent(const unsigned short* __restrict__ Xbf,
                const unsigned short* __restrict__ WihF, const unsigned short* __restrict__ WhhF,
                const unsigned short* __restrict__ WihB, const unsigned short* __restrict__ WhhB,
                const float* __restrict__ biasF, const float* __restrict__ biasB,
                unsigned short* __restrict__ Ycat)
{
    constexpr int Kx = C / 32;

    const int tid    = threadIdx.x;
    const int lane   = tid & 31;
    const int wave   = tid >> 5;          // 0..7
    const int mblock = blockIdx.x & 3;    // batch tile
    const int dir    = blockIdx.x >> 2;   // 0 = fwd, 1 = bwd
    const int mbase  = mblock * 16;
    const int hg     = wave;              // hidden col group (16 cols)
    const int ncol   = lane & 15;
    const int hi     = (lane >> 4) & 1;

    const unsigned short* Wih  = dir ? WihB  : WihF;
    const unsigned short* Whh  = dir ? WhhB  : WhhF;
    const float*          bias = dir ? biasB : biasF;

    unsigned short* sWih = lds_lstm;                   // GATES*C
    unsigned short* sH   = sWih + GATES * C;           // 2 * 16 * HID

    // --- cooperative fill of LDS (Wih only) --------------------------------
    {
        const int n0 = GATES * C / 8;                  // uint4 count
        const uint4* s0 = (const uint4*)Wih;
        uint4* d0 = (uint4*)sWih;
        for (int i = tid; i < n0; i += 256) d0[i] = s0[i];
        for (int i = tid; i < 2 * 16 * HID; i += 256) sH[i] = 0;   // h0 = 0
    }

    // --- Whh fragments -> registers for the whole scan ---------------------
    v16bf whI[4], whF[4], whG[4], whO[4];
    #pragma unroll
    for (int k = 0; k < 4; ++k) {
        whI[k] = load_frag(Whh + (  0 + hg * 16) * HID + k * 32, HID, lane);
        whF[k] = load_frag(Whh + (128 + hg * 16) * HID + k * 32, HID, lane);
        whG[k] = load_frag(Whh + (256 + hg * 16) * HID + k * 32, HID, lane);
        whO[k] = load_frag(Whh + (384 + hg * 16) * HID + k * 32, HID, lane);
    }

    // --- persistent bias tiles: C operand of each gate's first WMMA --------
    // (column = nbase + ncol, same value for both lane halves / all 8 rows)
    v8f btI, btF, btG, btO;
    {
        const float bI = bias[  0 + hg * 16 + ncol];
        const float bF = bias[128 + hg * 16 + ncol];
        const float bG = bias[256 + hg * 16 + ncol];
        const float bO = bias[384 + hg * 16 + ncol];
        #pragma unroll
        for (int r = 0; r < 8; ++r) { btI[r] = bI; btF[r] = bF; btG[r] = bG; btO[r] = bO; }
    }

    v8f cc;                                   // cell state tile, f32 in VGPRs
    #pragma unroll
    for (int r = 0; r < 8; ++r) cc[r] = 0.0f;

    // strength-reduced walking pointers (direction-signed strides)
    const long xstep = dir ? -(long)(NBATCH * C)  : (long)(NBATCH * C);
    const long ystep = dir ? -(long)(NBATCH * 256) : (long)(NBATCH * 256);
    const int  t0    = dir ? (T_LEN - 1) : 0;
    const unsigned short* xptr = Xbf + ((size_t)t0 * NBATCH + mbase) * C;
    unsigned short*       yptr = Ycat + (size_t)t0 * NBATCH * 256
                                      + (size_t)dir * 128 + hg * 16 + ncol;

    __syncthreads();

    int rbuf = 0;
    for (int t = 0; t < T_LEN; ++t) {
        const unsigned short* hRead  = sH + rbuf * 16 * HID;
        unsigned short*       hWrite = sH + (rbuf ^ 1) * 16 * HID;

        __builtin_prefetch(xptr + xstep + ncol * C, 0, 0);  // next step's x rows

        // input projection: gates = bias + x_t[16xC] * Wih^T  (A global, B LDS)
        v16bf a0 = load_frag(xptr, C, lane);
        v8f aI = wmma_bf16(a0, load_frag(sWih + (  0 + hg * 16) * C, C, lane), btI);
        v8f aF = wmma_bf16(a0, load_frag(sWih + (128 + hg * 16) * C, C, lane), btF);
        v8f aG = wmma_bf16(a0, load_frag(sWih + (256 + hg * 16) * C, C, lane), btG);
        v8f aO = wmma_bf16(a0, load_frag(sWih + (384 + hg * 16) * C, C, lane), btO);
        #pragma unroll
        for (int k = 1; k < Kx; ++k) {
            v16bf a = load_frag(xptr + k * 32, C, lane);
            aI = wmma_bf16(a, load_frag(sWih + (  0 + hg * 16) * C + k * 32, C, lane), aI);
            aF = wmma_bf16(a, load_frag(sWih + (128 + hg * 16) * C + k * 32, C, lane), aF);
            aG = wmma_bf16(a, load_frag(sWih + (256 + hg * 16) * C + k * 32, C, lane), aG);
            aO = wmma_bf16(a, load_frag(sWih + (384 + hg * 16) * C + k * 32, C, lane), aO);
        }
        // recurrent projection: gates += h[16x128] * Whh^T  (A LDS, B registers)
        #pragma unroll
        for (int k = 0; k < HID / 32; ++k) {
            v16bf a = load_frag(hRead + k * 32, HID, lane);
            aI = wmma_bf16(a, whI[k], aI);
            aF = wmma_bf16(a, whF[k], aF);
            aG = wmma_bf16(a, whG[k], aG);
            aO = wmma_bf16(a, whO[k], aO);
        }

        // pointwise LSTM cell update (tile layout: elem r -> row r+hi*8, col ncol)
        #pragma unroll
        for (int r = 0; r < 8; ++r) {
            float ig = hsig(aI[r]);
            float fg = hsig(aF[r]);
            float gg = tanhf(aG[r]);
            float og = hsig(aO[r]);
            float c  = fg * cc[r] + ig * gg;
            cc[r] = c;
            float h = og * tanhf(c);
            unsigned short hb = f2bf(h);
            int mloc = r + hi * 8;
            hWrite[mloc * HID + hg * 16 + ncol] = hb;
            yptr[(size_t)(mbase + mloc) * 256] = hb;
        }

        xptr += xstep;
        yptr += ystep;
        rbuf ^= 1;
        __syncthreads();   // h(t) visible to all waves before step t+1
    }
}

// ---------------------------------------------------------------------------
// 1x1 conv as GEMM: Z[TB][128] = Ycat[TB][256](bf16) * convW[128][256]^T + convB
// one wave per 16-row M tile; conv_w staged in LDS (64KB).
__global__ void __launch_bounds__(256)
conv_gemm(const unsigned short* __restrict__ Ycat,
          const unsigned short* __restrict__ convW,   // [128][256] bf16
          const float* __restrict__ convB,
          float* __restrict__ Z)
{
    __shared__ unsigned short sCW[128 * 256];
    {
        const uint4* s = (const uint4*)convW;
        uint4* d = (uint4*)sCW;
        for (int i = threadIdx.x; i < (128 * 256) / 8; i += 256) d[i] = s[i];
    }
    __syncthreads();

    const int lane  = threadIdx.x & 31;
    const int wave  = threadIdx.x >> 5;
    const int mtile = blockIdx.x * 8 + wave;            // grid sized exactly
    const int ncol  = lane & 15;
    const int hi    = (lane >> 4) & 1;

    const unsigned short* arow = Ycat + (size_t)mtile * 16 * 256;
    v16bf afr[8];
    #pragma unroll
    for (int k = 0; k < 8; ++k) afr[k] = load_frag(arow + k * 32, 256, lane);

    #pragma unroll
    for (int n = 0; n < 8; ++n) {
        float bb = convB[n * 16 + ncol];
        v8f acc;
        #pragma unroll
        for (int r = 0; r < 8; ++r) acc[r] = bb;
        #pragma unroll
        for (int k = 0; k < 8; ++k)
            acc = wmma_bf16(afr[k], load_frag(sCW + (n * 16) * 256 + k * 32, 256, lane), acc);
        #pragma unroll
        for (int r = 0; r < 8; ++r)
            Z[((size_t)mtile * 16 + r + hi * 8) * 128 + n * 16 + ncol] = acc[r];
    }
}

// ---------------------------------------------------------------------------
// BatchNorm: deterministic two-stage reduction over (B,T) per channel.
__global__ void bn_stage1(const float* __restrict__ Z,
                          float* __restrict__ partS, float* __restrict__ partQ) {
    const int ch = threadIdx.x;           // 128 threads
    float s = 0.0f, q = 0.0f;
    for (int r = blockIdx.x; r < TBROWS; r += 256) {   // gridDim.x == 256
        float v = Z[(size_t)r * 128 + ch];
        s += v;
        q += v * v;
    }
    partS[blockIdx.x * 128 + ch] = s;
    partQ[blockIdx.x * 128 + ch] = q;
}

__global__ void bn_stage2(const float* __restrict__ partS, const float* __restrict__ partQ,
                          float* __restrict__ stats) {
    const int tid = threadIdx.x;          // 256 threads
    if (tid < 128) {
        float s = 0.0f;
        for (int b = 0; b < 256; ++b) s += partS[b * 128 + tid];
        stats[tid] = s;
    } else {
        const int ch = tid - 128;
        float q = 0.0f;
        for (int b = 0; b < 256; ++b) q += partQ[b * 128 + ch];
        stats[128 + ch] = q;
    }
}

__global__ void bn_apply(const float* __restrict__ Z, const float* __restrict__ stats,
                         const float* __restrict__ gamma, const float* __restrict__ beta,
                         unsigned short* __restrict__ Xbf, float* __restrict__ out, int last) {
    size_t i = (size_t)blockIdx.x * blockDim.x + threadIdx.x;   // < TBROWS*128
    const int ch = (int)(i & 127);
    const size_t row = i >> 7;
    const float invN = 1.0f / (float)TBROWS;
    const float mean = stats[ch] * invN;
    const float var  = stats[128 + ch] * invN - mean * mean;
    const float sc   = gamma[ch] * rsqrtf(var + 1e-3f);
    const float v    = (Z[i] - mean) * sc + beta[ch];
    if (last) {
        const int b = (int)(row & 63);
        const int t = (int)(row >> 6);
        out[((size_t)b * 128 + ch) * T_LEN + t] = v;     // [B, H, T]
    } else {
        Xbf[i] = f2bf(v);                                // next block input [T*B, 128]
    }
}

// ---------------------------------------------------------------------------
extern "C" void kernel_launch(void* const* d_in, const int* in_sizes, int n_in,
                              void* d_out, int out_size, void* d_ws, size_t ws_size,
                              hipStream_t stream) {
    (void)in_sizes; (void)n_in; (void)out_size; (void)ws_size;
    const float* x = (const float*)d_in[0];

    char* w = (char*)d_ws;
    auto carve = [&](size_t bytes) -> void* {
        void* p = (void*)w;
        w += (bytes + 255) & ~(size_t)255;
        return p;
    };
    unsigned short* Xbf  = (unsigned short*)carve((size_t)TBROWS * 128 * 2);
    unsigned short* Ycat = (unsigned short*)carve((size_t)TBROWS * 256 * 2);
    float*          Z    = (float*)carve((size_t)TBROWS * 128 * 4);
    unsigned short* wihF = (unsigned short*)carve(512 * 128 * 2);
    unsigned short* whhF = (unsigned short*)carve(512 * 128 * 2);
    unsigned short* wihB = (unsigned short*)carve(512 * 128 * 2);
    unsigned short* whhB = (unsigned short*)carve(512 * 128 * 2);
    unsigned short* cW   = (unsigned short*)carve(128 * 256 * 2);
    float* biasF = (float*)carve(512 * 4);
    float* biasB = (float*)carve(512 * 4);
    float* partS = (float*)carve(256 * 128 * 4);
    float* partQ = (float*)carve(256 * 128 * 4);
    float* stats = (float*)carve(256 * 4);

    transform_x<<<48000, 256, 0, stream>>>(x, Xbf);

    for (int bi = 0; bi < 3; ++bi) {
        const int C = bi ? 128 : 64;
        const float* const* P = (const float* const*)(d_in + 1 + bi * 12);
        // P: 0 wih_f, 1 whh_f, 2 bih_f, 3 bhh_f, 4 wih_b, 5 whh_b, 6 bih_b,
        //    7 bhh_b, 8 conv_w, 9 conv_b, 10 gamma, 11 beta
        cvt_bf16<<<(512 * C + 255) / 256, 256, 0, stream>>>(P[0], wihF, 512 * C);
        cvt_bf16<<<(512 * 128 + 255) / 256, 256, 0, stream>>>(P[1], whhF, 512 * 128);
        cvt_bf16<<<(512 * C + 255) / 256, 256, 0, stream>>>(P[4], wihB, 512 * C);
        cvt_bf16<<<(512 * 128 + 255) / 256, 256, 0, stream>>>(P[5], whhB, 512 * 128);
        cvt_bf16<<<(128 * 256 + 255) / 256, 256, 0, stream>>>(P[8], cW, 128 * 256);
        add_f32<<<2, 256, 0, stream>>>(P[2], P[3], biasF, 512);
        add_f32<<<2, 256, 0, stream>>>(P[6], P[7], biasB, 512);

        const size_t smem = ((size_t)GATES * C + 2 * 16 * HID) * 2;
        if (bi == 0) {
            hipFuncSetAttribute((const void*)lstm_persistent<64>,
                                hipFuncAttributeMaxDynamicSharedMemorySize, (int)smem);
            lstm_persistent<64><<<8, 256, smem, stream>>>(Xbf, wihF, whhF, wihB, whhB,
                                                          biasF, biasB, Ycat);
        } else {
            hipFuncSetAttribute((const void*)lstm_persistent<128>,
                                hipFuncAttributeMaxDynamicSharedMemorySize, (int)smem);
            lstm_persistent<128><<<8, 256, smem, stream>>>(Xbf, wihF, whhF, wihB, whhB,
                                                           biasF, biasB, Ycat);
        }

        conv_gemm<<<MTILES / 8, 256, 0, stream>>>(Ycat, cW, P[9], Z);

        bn_stage1<<<256, 128, 0, stream>>>(Z, partS, partQ);
        bn_stage2<<<1, 256, 0, stream>>>(partS, partQ, stats);
        bn_apply<<<(TBROWS * 128) / 256, 256, 0, stream>>>(Z, stats, P[10], P[11],
                                                           Xbf, (float*)d_out, bi == 2 ? 1 : 0);
    }
}
